// Encoder_65335042506817
// MI455X (gfx1250) — compile-verified, matrix operand order checked by reading, not run on hardware
//
#include <hip/hip_runtime.h>
#include <hip/hip_bf16.h>

// =====================================================================
// CDNA5 (gfx1250) implementation of the 2-block IPA protein encoder.
// All dense GEMMs run on v_wmma_f32_16x16x32_bf16 (bf16 in, fp32 acc),
// with A-operands held in registers and B-operands streamed from a
// pre-packed bf16 tile layout (two global_load_b128 per fragment).
// Tile staging from HBM uses GLOBAL_LOAD_ASYNC_TO_LDS_B128 (ASYNCcnt).
// Irregular neighbor work (K=32 gathers, softmax, frames) stays VALU fp32.
// =====================================================================

#define DEV static __device__ __forceinline__

typedef __attribute__((ext_vector_type(16))) __bf16 bf16x16;
typedef __attribute__((ext_vector_type(2)))  __bf16 bf16x2;
typedef __attribute__((ext_vector_type(8)))  float  f32x8;

constexpr int NRES = 8192, KNB = 32, NA = 5, D = 256, P = 64, H = 8, KD = 32,
              NP = 4, FF = 512;
constexpr int FEATP = 96;     // A*GEOM_F = 95 padded to 96
constexpr int OUTF  = 896;    // H*(KD+P+NP*3+NP)
constexpr int TR    = 16;     // residues per workgroup (one WMMA row-tile)
constexpr int TPB   = 128;    // 4 wave32

// ---------------- bf16 helpers (bit-exact, no reliance on __bf16 arith) ----
DEV __bf16 f2bf(float f) {
  union { float f; unsigned u; } x; x.f = f;
  unsigned r = (x.u + 0x7fffu + ((x.u >> 16) & 1u)) >> 16;
  union { unsigned short s; __bf16 b; } y; y.s = (unsigned short)r;
  return y.b;
}
DEV float bf2f(__bf16 b) {
  union { unsigned short s; __bf16 b; } y; y.b = b;
  union { unsigned u; float f; } x; x.u = ((unsigned)y.s) << 16;
  return x.f;
}

// ---------------- async global->LDS staging (CDNA5 ASYNCcnt path) ----------
DEV void async_load_f32(float* ldsdst, const float* gsrc, int nfloats, int tid) {
  int nops = nfloats >> 2;  // 16B per op
  for (int i = tid; i < nops; i += TPB) {
    unsigned lo = (unsigned)(size_t)(const void*)(ldsdst + i * 4);
    const float* ga = gsrc + (size_t)i * 4;
    asm volatile("global_load_async_to_lds_b128 %0, %1, off"
                 :: "v"(lo), "v"(ga) : "memory");
  }
}
DEV void async_wait() { asm volatile("s_wait_asynccnt 0x0" ::: "memory"); }

// ---------------- WMMA fragment helpers (wave32 layouts per ISA 7.12.2) ----
// A (16x32 bf16): lane m=L&15, grp=L>>4; VGPR i holds K = grp*8 + (i>=4?16:0) + (i&3)*2 {+0,+1}
DEV bf16x16 fragA(const __bf16* X, int ldx, int lane, int k0) {
  int m = lane & 15, grp = lane >> 4;
  const __bf16* row = X + m * ldx + k0 + grp * 8;
  bf16x16 a;
#pragma unroll
  for (int i = 0; i < 8; ++i) {
    int kk = ((i & 4) ? 16 : 0) + ((i & 3) << 1);
    bf16x2 p = *(const bf16x2*)(row + kk);
    a[2 * i] = p[0]; a[2 * i + 1] = p[1];
  }
  return a;
}
// B tiles are pre-packed so a lane's 16 values are contiguous (two b128 loads)
DEV bf16x16 fragB(const __bf16* Wp, int ktiles, int kt, int nt, int lane) {
  return *(const bf16x16*)(Wp + (((size_t)(nt * ktiles + kt)) << 9) + (lane << 4));
}
// C/D (16x16 f32): VGPR r, lanes0-15 -> M=r, lanes16-31 -> M=8+r; N = lane&15
DEV f32x8 fragCload(const float* C, int ldc, int lane) {
  int n = lane & 15, mo = (lane >> 4) * 8;
  f32x8 c;
#pragma unroll
  for (int r = 0; r < 8; ++r) c[r] = C[(mo + r) * ldc + n];
  return c;
}
DEV void fragCstore(float* C, int ldc, int lane, f32x8 c) {
  int n = lane & 15, mo = (lane >> 4) * 8;
#pragma unroll
  for (int r = 0; r < 8; ++r) C[(mo + r) * ldc + n] = c[r];
}

// GEMM with A fragments resident in registers (reused across n-tiles).
template <int KT>
DEV f32x8 gemm_reg(const bf16x16 (&af)[KT], const __bf16* Wp, int nt, int lane,
                   f32x8 c) {
#pragma unroll
  for (int kt = 0; kt < KT; ++kt) {
    if (kt + 1 < KT)  // global_prefetch_b8 on next weight tile, keep near
      __builtin_prefetch((const void*)(Wp + (((size_t)(nt * KT + kt + 1)) << 9)), 0, 3);
    bf16x16 b = fragB(Wp, KT, kt, nt, lane);
    c = __builtin_amdgcn_wmma_f32_16x16x32_bf16(false, af[kt], false, b,
                                                (short)0, c, false, false);
  }
  return c;
}

// GEMM with A streamed from LDS (for large-K where A can't stay resident).
DEV f32x8 gemm_acc(f32x8 c, const __bf16* A, int lda, const __bf16* Wp,
                   int ktiles, int nt, int lane) {
#pragma unroll 4
  for (int kt = 0; kt < ktiles; ++kt) {
    if (kt + 1 < ktiles)
      __builtin_prefetch((const void*)(Wp + (((size_t)(nt * ktiles + kt + 1)) << 9)), 0, 3);
    bf16x16 a = fragA(A, lda, lane, kt * 32);
    bf16x16 b = fragB(Wp, ktiles, kt, nt, lane);
    c = __builtin_amdgcn_wmma_f32_16x16x32_bf16(false, a, false, b, (short)0, c,
                                                false, false);
  }
  return c;
}

// ---------------- misc math ------------------------------------------------
DEV float geluf(float x) {
  float x3 = x * x * x;
  return 0.5f * x * (1.f + tanhf(0.7978845608f * (x + 0.044715f * x3)));
}
DEV float softplusf(float x) { return (x > 20.f) ? x : log1pf(__expf(x)); }

// frames: Rc[0..2]=e1, [3..5]=e2, [6..8]=e3 (columns), [9..11]=t(=CA)
DEV void frames_calc(const float* p, float* Rc) {
  float n0 = p[0], n1 = p[1], n2 = p[2];
  float a0 = p[3], a1 = p[4], a2 = p[5];
  float c0 = p[6], c1 = p[7], c2 = p[8];
  float e0 = c0 - a0, e1 = c1 - a1, e2 = c2 - a2;
  float inv = rsqrtf(e0 * e0 + e1 * e1 + e2 * e2 + 1e-12f);
  e0 *= inv; e1 *= inv; e2 *= inv;
  float u0 = n0 - a0, u1 = n1 - a1, u2 = n2 - a2;
  float du = u0 * e0 + u1 * e1 + u2 * e2;
  float v0 = u0 - du * e0, v1 = u1 - du * e1, v2 = u2 - du * e2;
  inv = rsqrtf(v0 * v0 + v1 * v1 + v2 * v2 + 1e-12f);
  v0 *= inv; v1 *= inv; v2 *= inv;
  float w0 = e1 * v2 - e2 * v1, w1 = e2 * v0 - e0 * v2, w2 = e0 * v1 - e1 * v0;
  Rc[0] = e0; Rc[1] = e1; Rc[2] = e2;
  Rc[3] = v0; Rc[4] = v1; Rc[5] = v2;
  Rc[6] = w0; Rc[7] = w1; Rc[8] = w2;
  Rc[9] = a0; Rc[10] = a1; Rc[11] = a2;
}

// LN over 16 rows of [16][256] LDS fp32 -> bf16. Called by all 128 threads.
DEV void ln_rows(const float* src, __bf16* dst, const float* g, const float* b,
                 float* red, float* stat, int tid) {
  int r = tid >> 3, seg = tid & 7;
  float s = 0.f, s2 = 0.f;
  const float* row = src + r * D + seg * 32;
  for (int i = 0; i < 32; ++i) { float v = row[i]; s += v; s2 += v * v; }
  red[(r * 8 + seg) * 2] = s; red[(r * 8 + seg) * 2 + 1] = s2;
  __syncthreads();
  if (tid < TR) {
    float S = 0.f, S2 = 0.f;
    for (int k = 0; k < 8; ++k) { S += red[(tid * 8 + k) * 2]; S2 += red[(tid * 8 + k) * 2 + 1]; }
    float m = S / (float)D, v = S2 / (float)D - m * m;
    stat[tid * 2] = m; stat[tid * 2 + 1] = rsqrtf(v + 1e-5f);
  }
  __syncthreads();
  for (int idx = tid; idx < TR * D; idx += TPB) {
    int rr = idx >> 8, c = idx & 255;
    dst[idx] = f2bf((src[idx] - stat[rr * 2]) * stat[rr * 2 + 1] * g[c] + b[c]);
  }
  __syncthreads();
}

// ---------------- parameter / workspace structs ----------------------------
struct WSB {
  float* loc; const float* posIn; float* posOut;
  float *Rw, *tw, *qw, *kw, *vw, *qpgw, *kpgw, *vpgw, *centw, *neiw;
  const int *nbr, *resi, *chain;
  const unsigned char *umask, *mask;
};
struct BWT {
  const float *b_p1, *gamma, *ln_a_b, *ln_a_g, *ln_m_b, *ln_m_g, *ln_p1_b,
      *ln_p1_g, *ln_p2_b, *ln_p2_g, *ln_u_b, *ln_u_g, *w_bias, *w_geom, *w_resi;
  const __bf16 *p_feat, *p_center, *p_nei, *p_p1, *p_p2, *p_q, *p_k, *p_v,
      *p_qp, *p_kp, *p_vp, *p_out, *p_g, *p_v2, *p_o, *p_pos;
};

// ---------------- weight packing ------------------------------------------
// Packs fp32 W[Ks][Ns] (row-major) into WMMA B-fragment tile order:
// Wp[(nt*ktiles+kt)*512 + lane*16 + j], zero-padded to KpxNp.
__global__ void pack_weight(const float* __restrict__ W, __bf16* __restrict__ Wp,
                            int Ks, int Ns, int Kp, int Np) {
  int ktiles = Kp / 32;
  int tid = blockIdx.x * blockDim.x + threadIdx.x;
  int total = ktiles * (Np / 16) * 512;
  if (tid >= total) return;
  int j = tid & 15, lane = (tid >> 4) & 31, tile = tid >> 9;
  int kt = tile % ktiles, nt = tile / ktiles;
  int n = lane & 15, grp = lane >> 4;
  int i = j >> 1, bb = j & 1;
  int kk = ((i < 4) ? 0 : 16) + grp * 8 + ((i & 3) << 1) + bb;
  int k = kt * 32 + kk, col = nt * 16 + n;
  float v = (k < Ks && col < Ns) ? W[(size_t)k * Ns + col] : 0.f;
  Wp[tid] = f2bf(v);
}

// ---------------- Stage A LDS layout ---------------------------------------
constexpr size_t SA_POSL = 0;                       // float[16][16]
constexpr size_t SA_RF   = SA_POSL + 1024;          // float[16][12]
constexpr size_t SA_FEAT = SA_RF + 768;             // bf16 [16][96]
constexpr size_t SA_LOCF = SA_FEAT + 16 * 96 * 2;   // float[16][256]
constexpr size_t SA_LB   = SA_LOCF + 16 * 256 * 4;  // bf16 [16][256]
constexpr size_t SA_XB   = SA_LB + 16 * 256 * 2;    // bf16 [16][256]
constexpr size_t SA_PQV  = SA_XB + 16 * 256 * 2;    // float[16][288] qp|kp|vp
constexpr size_t SA_RED  = SA_PQV + 16 * 288 * 4;   // float[256]
constexpr size_t SA_STAT = SA_RED + 1024;           // float[32]
constexpr size_t SA_SZ   = SA_STAT + 128;

__global__ __launch_bounds__(TPB) void stageA(WSB ws, BWT bw) {
  const int tid = threadIdx.x, lane = tid & 31, wv = tid >> 5;
  const int rt = blockIdx.x * TR;
  extern __shared__ char smem[];
  float*  posl  = (float*)(smem + SA_POSL);
  float*  Rf    = (float*)(smem + SA_RF);
  __bf16* featb = (__bf16*)(smem + SA_FEAT);
  float*  locf  = (float*)(smem + SA_LOCF);
  __bf16* lb    = (__bf16*)(smem + SA_LB);
  __bf16* xb    = (__bf16*)(smem + SA_XB);
  float*  pqv   = (float*)(smem + SA_PQV);
  float*  red   = (float*)(smem + SA_RED);
  float*  stat  = (float*)(smem + SA_STAT);

  // async-stage the local tile while we do scalar setup
  async_load_f32(locf, ws.loc + (size_t)rt * D, TR * D, tid);
  for (int i = tid; i < TR * 15; i += TPB) {
    int r = i / 15, c = i % 15;
    posl[r * 16 + c] = ws.posIn[(size_t)rt * 15 + i];
  }
  async_wait();
  __syncthreads();

  if (tid < TR) {
    frames_calc(posl + tid * 16, Rf + tid * 12);
    for (int j = 0; j < 9; ++j) ws.Rw[(size_t)(rt + tid) * 9 + j] = Rf[tid * 12 + j];
    for (int j = 0; j < 3; ++j) ws.tw[(size_t)(rt + tid) * 3 + j] = Rf[tid * 12 + 9 + j];
  }
  __syncthreads();

  // geometric features: [unit(lp) (A*3) | rbf(|lp|,10,16) (A*16)]
  if (tid < TR * NA) {
    int r = tid / NA, a = tid % NA;
    const float* pp = posl + r * 16 + a * 3;
    const float* Rc = Rf + r * 12;
    float d0 = pp[0] - Rc[9], d1 = pp[1] - Rc[10], d2 = pp[2] - Rc[11];
    float l0 = Rc[0] * d0 + Rc[1] * d1 + Rc[2] * d2;
    float l1 = Rc[3] * d0 + Rc[4] * d1 + Rc[5] * d2;
    float l2 = Rc[6] * d0 + Rc[7] * d1 + Rc[8] * d2;
    float nn = sqrtf(l0 * l0 + l1 * l1 + l2 * l2 + 1e-12f);
    float inv = 1.f / nn;
    __bf16* fr = featb + r * FEATP;
    fr[a * 3 + 0] = f2bf(l0 * inv); fr[a * 3 + 1] = f2bf(l1 * inv); fr[a * 3 + 2] = f2bf(l2 * inv);
    for (int bq = 0; bq < 16; ++bq) {
      float z = (nn - (float)bq * (10.f / 15.f)) / 0.625f;
      fr[15 + a * 16 + bq] = f2bf(__expf(-z * z));
    }
  }
  if (tid >= TR * NA && tid < TR * NA + TR) featb[(tid - TR * NA) * FEATP + 95] = f2bf(0.f);
  __syncthreads();

  // local += feat @ w_feat  (A fragments resident in registers)
  {
    bf16x16 af[3];
#pragma unroll
    for (int kt = 0; kt < 3; ++kt) af[kt] = fragA(featb, FEATP, lane, kt * 32);
    for (int nt = wv; nt < 16; nt += 4) {
      f32x8 c = fragCload(locf + nt * 16, D, lane);
      c = gemm_reg<3>(af, bw.p_feat, nt, lane, c);
      fragCstore(locf + nt * 16, D, lane, c);
    }
  }
  __syncthreads();

  for (int i = tid; i < TR * D; i += TPB) {
    float v = locf[i];
    ws.loc[(size_t)rt * D + i] = v;
    lb[i] = f2bf(v);
  }
  __syncthreads();
  ln_rows(locf, xb, bw.ln_a_g, bw.ln_a_b, red, stat, tid);

  // 66 ntile units from x=LN(local): q(16) k(16) v(16) qp(6) kp(6) vp(6)
  {
    bf16x16 af[8];
#pragma unroll
    for (int kt = 0; kt < 8; ++kt) af[kt] = fragA(xb, D, lane, kt * 32);
    for (int u = wv; u < 66; u += 4) {
      const __bf16* Wp; int nt; float* gdst = nullptr; float* ldst = nullptr;
      if (u < 48) {
        int jj = u >> 4; nt = u & 15;
        Wp   = (jj == 0) ? bw.p_q : ((jj == 1) ? bw.p_k : bw.p_v);
        gdst = (jj == 0) ? ws.qw : ((jj == 1) ? ws.kw : ws.vw);
      } else {
        int uu = u - 48; int jj = uu / 6; nt = uu % 6;
        Wp = (jj == 0) ? bw.p_qp : ((jj == 1) ? bw.p_kp : bw.p_vp);
        ldst = pqv + jj * 96;
      }
      f32x8 c = {};
      c = gemm_reg<8>(af, Wp, nt, lane, c);
      if (gdst) {
        int n = lane & 15, mo = (lane >> 4) * 8;
#pragma unroll
        for (int r = 0; r < 8; ++r)
          gdst[(size_t)(rt + mo + r) * 256 + nt * 16 + n] = c[r];
      } else {
        fragCstore(ldst + nt * 16, 288, lane, c);
      }
    }
  }
  // 8 ntile units from local: center(4) nei(4)
  {
    bf16x16 af[8];
#pragma unroll
    for (int kt = 0; kt < 8; ++kt) af[kt] = fragA(lb, D, lane, kt * 32);
    for (int u = 66 + wv; u < 74; u += 4) {
      int uu = u - 66, jj = uu >> 2, nt = uu & 3;
      const __bf16* Wp = (jj == 0) ? bw.p_center : bw.p_nei;
      float* gdst = (jj == 0) ? ws.centw : ws.neiw;
      f32x8 c = {};
      c = gemm_reg<8>(af, Wp, nt, lane, c);
      int n = lane & 15, mo = (lane >> 4) * 8;
#pragma unroll
      for (int r = 0; r < 8; ++r)
        gdst[(size_t)(rt + mo + r) * 64 + nt * 16 + n] = c[r];
    }
  }
  __syncthreads();

  // rotate qp/kp/vp into global frame, spill for Stage B's neighbor gathers
  for (int u = tid; u < TR * 32; u += TPB) {
    int r = u >> 5, hp = u & 31;
    const float* Rc = Rf + r * 12;
    float t0 = Rc[9], t1 = Rc[10], t2 = Rc[11];
    for (int kind = 0; kind < 3; ++kind) {
      const float* src = pqv + r * 288 + kind * 96 + hp * 3;
      float a0 = src[0], a1 = src[1], a2 = src[2];
      float g0 = Rc[0] * a0 + Rc[3] * a1 + Rc[6] * a2 + t0;
      float g1 = Rc[1] * a0 + Rc[4] * a1 + Rc[7] * a2 + t1;
      float g2 = Rc[2] * a0 + Rc[5] * a1 + Rc[8] * a2 + t2;
      float* dst = ((kind == 0) ? ws.qpgw : (kind == 1) ? ws.kpgw : ws.vpgw) +
                   (size_t)(rt + r) * 96 + hp * 3;
      dst[0] = g0; dst[1] = g1; dst[2] = g2;
    }
  }
}

// ---------------- Stage B LDS layout ---------------------------------------
constexpr size_t SB_NBL  = 0;                 // int  [16][32]
constexpr size_t SB_MB   = SB_NBL + 2048;     // float[16][32]
constexpr size_t SB_RF   = SB_MB + 2048;      // float[16][12]
constexpr size_t SB_CAL  = SB_RF + 768;       // float[16][4]
constexpr size_t SB_RES  = SB_CAL + 256;      // int[16]
constexpr size_t SB_CHN  = SB_RES + 64;       // int[16]
constexpr size_t SB_CEN  = SB_CHN + 64;       // float[16][64]
constexpr size_t SB_RED  = SB_CEN + 4096;     // float[256]
constexpr size_t SB_STAT = SB_RED + 1024;     // float[32]
constexpr size_t SB_UPD  = SB_STAT + 128;     // float[16][16]
constexpr size_t SB_PAIR = SB_UPD + 1024;     // bf16 [512][64]
constexpr size_t SB_SCR  = SB_PAIR + 65536;   // 32KB phase scratch
constexpr size_t SB_OUT  = SB_SCR + 32768;    // bf16 [16][896]
constexpr size_t SB_LOCF = SB_OUT + 28672;    // float[16][256]
constexpr size_t SB_SZ   = SB_LOCF + 16384;   // 154880 B (< 160KB; CDNA5 WGP has 320KB)

__global__ __launch_bounds__(TPB) void stageB(WSB ws, BWT bw) {
  const int tid = threadIdx.x, lane = tid & 31, wv = tid >> 5;
  const int rt = blockIdx.x * TR;
  extern __shared__ char smem[];
  int*    nbl   = (int*)(smem + SB_NBL);
  float*  mb    = (float*)(smem + SB_MB);
  float*  Rf    = (float*)(smem + SB_RF);
  float*  cal   = (float*)(smem + SB_CAL);
  int*    resl  = (int*)(smem + SB_RES);
  int*    chnl  = (int*)(smem + SB_CHN);
  float*  centl = (float*)(smem + SB_CEN);
  float*  red   = (float*)(smem + SB_RED);
  float*  stat  = (float*)(smem + SB_STAT);
  float*  updl  = (float*)(smem + SB_UPD);
  __bf16* pairb = (__bf16*)(smem + SB_PAIR);
  char*   scr   = smem + SB_SCR;
  __bf16* outb  = (__bf16*)(smem + SB_OUT);
  float*  locf  = (float*)(smem + SB_LOCF);

  // ---- phase 0: async-stage local & center, gather per-tile state ----
  async_load_f32(locf, ws.loc + (size_t)rt * D, TR * D, tid);
  async_load_f32(centl, ws.centw + (size_t)rt * P, TR * P, tid);
  for (int i = tid; i < TR * KNB; i += TPB) {
    int g = ws.nbr[(size_t)rt * KNB + i];
    int cl = (g < 0) ? 0 : g;
    nbl[i] = cl;
    mb[i] = (g >= 0 && ws.mask[cl]) ? 0.f : -1e9f;
  }
  if (tid < TR) {
    for (int j = 0; j < 9; ++j) Rf[tid * 12 + j] = ws.Rw[(size_t)(rt + tid) * 9 + j];
    for (int j = 0; j < 3; ++j) Rf[tid * 12 + 9 + j] = ws.tw[(size_t)(rt + tid) * 3 + j];
    cal[tid * 4 + 0] = ws.posIn[(size_t)(rt + tid) * 15 + 3];
    cal[tid * 4 + 1] = ws.posIn[(size_t)(rt + tid) * 15 + 4];
    cal[tid * 4 + 2] = ws.posIn[(size_t)(rt + tid) * 15 + 5];
    resl[tid] = ws.resi[rt + tid];
    chnl[tid] = ws.chain[rt + tid];
  }
  async_wait();
  __syncthreads();

  // ---- phase 1: pair features + LN1 (VALU, 4 rows/thread) ------------
  for (int row = tid; row < TR * KNB; row += TPB) {
    int r = row >> 5;
    int nbj = nbl[row];
    int dr = resl[r] - ws.resi[nbj];
    dr = (dr < -32) ? -32 : ((dr > 32) ? 32 : dr);
    dr += 32;
    if (chnl[r] != ws.chain[nbj]) dr = 65;
    const float* wres = bw.w_resi + dr * P;
    const float* Rc = Rf + r * 12;
    float d0 = ws.posIn[(size_t)nbj * 15 + 3] - cal[r * 4 + 0];
    float d1 = ws.posIn[(size_t)nbj * 15 + 4] - cal[r * 4 + 1];
    float d2 = ws.posIn[(size_t)nbj * 15 + 5] - cal[r * 4 + 2];
    float g0 = Rc[0] * d0 + Rc[1] * d1 + Rc[2] * d2;
    float g1 = Rc[3] * d0 + Rc[4] * d1 + Rc[5] * d2;
    float g2 = Rc[6] * d0 + Rc[7] * d1 + Rc[8] * d2;
    float dn = sqrtf(g0 * g0 + g1 * g1 + g2 * g2 + 1e-12f);
    float geom[19];
    float invd = 1.f / (dn + 1e-8f);
    geom[0] = g0 * invd; geom[1] = g1 * invd; geom[2] = g2 * invd;
    for (int bq = 0; bq < 16; ++bq) {
      float z = (dn - (float)bq * (22.f / 15.f)) / 1.375f;
      geom[3 + bq] = __expf(-z * z);
    }
    float acc[64];
    const float* neirow = ws.neiw + (size_t)nbj * P;
    for (int c = 0; c < P; ++c) acc[c] = wres[c] + centl[r * P + c] + neirow[c];
    for (int gg = 0; gg < 19; ++gg) {
      float gv = geom[gg];
      const float* wg = bw.w_geom + gg * P;
      for (int c = 0; c < P; ++c) acc[c] += gv * wg[c];
    }
    float s = 0.f, s2 = 0.f;
    for (int c = 0; c < P; ++c) { s += acc[c]; s2 += acc[c] * acc[c]; }
    float m = s / (float)P, var = s2 / (float)P - m * m, rs = rsqrtf(var + 1e-5f);
    __bf16* pr = pairb + row * P;
    for (int c = 0; c < P; ++c)
      pr[c] = f2bf((acc[c] - m) * rs * bw.ln_p1_g[c] + bw.ln_p1_b[c]);
  }
  __syncthreads();

  // ---- phase 1b: pair MLP 64->128(gelu)->64 + LN2 (WMMA, per-wave) ---
  {
    __bf16* hb = (__bf16*)scr + wv * 2048;            // [16][128] per wave
    float*  p2 = (float*)(scr + 16384) + wv * 1024;   // [16][64]  per wave
    float*  ms = red + wv * 32;
    for (int mt = wv; mt < 32; mt += 4) {
      const __bf16* Xm = pairb + mt * 16 * P;
      bf16x16 afp[2];
#pragma unroll
      for (int kt = 0; kt < 2; ++kt) afp[kt] = fragA(Xm, P, lane, kt * 32);
      for (int nt = 0; nt < 8; ++nt) {
        f32x8 c = {};
        c = gemm_reg<2>(afp, bw.p_p1, nt, lane, c);
        int n = lane & 15, mo = (lane >> 4) * 8;
        float bb = bw.b_p1[nt * 16 + n];
#pragma unroll
        for (int r = 0; r < 8; ++r)
          hb[(mo + r) * 128 + nt * 16 + n] = f2bf(geluf(c[r] + bb));
      }
      bf16x16 afh[4];
#pragma unroll
      for (int kt = 0; kt < 4; ++kt) afh[kt] = fragA(hb, 128, lane, kt * 32);
      for (int nt = 0; nt < 4; ++nt) {
        f32x8 c = {};
        c = gemm_reg<4>(afh, bw.p_p2, nt, lane, c);
        fragCstore(p2 + nt * 16, P, lane, c);
      }
      if (lane < 16) {  // per-row stats (same-wave LDS ops are in-order)
        float s = 0.f, s2 = 0.f;
        const float* pr = p2 + lane * P;
        for (int c = 0; c < P; ++c) { s += pr[c]; s2 += pr[c] * pr[c]; }
        float m = s / (float)P, var = s2 / (float)P - m * m;
        ms[lane * 2] = m; ms[lane * 2 + 1] = rsqrtf(var + 1e-5f);
      }
      int row = lane & 15, half = lane >> 4;
      float m = ms[row * 2], rs = ms[row * 2 + 1];
      __bf16* dst = pairb + (mt * 16 + row) * P + half * 32;
      const float* srcr = p2 + row * P + half * 32;
      for (int c = 0; c < 32; ++c)
        dst[c] = f2bf((srcr[c] - m) * rs * bw.ln_p2_g[half * 32 + c] +
                      bw.ln_p2_b[half * 32 + c]);
    }
  }
  __syncthreads();

  // ---- phase 2: IPA attention, one thread per (residue, head) --------
  {
    int r = tid >> 3, h = tid & 7;
    float q[KD];
    const float* qg = ws.qw + (size_t)(rt + r) * 256 + h * KD;
    for (int d = 0; d < KD; ++d) q[d] = qg[d];
    float qp[12];
    const float* qpg = ws.qpgw + (size_t)(rt + r) * 96 + h * 12;
    for (int i = 0; i < 12; ++i) qp[i] = qpg[i];
    float sp = softplusf(bw.gamma[h]);
    float logit[KNB], mx = -1e30f;
    for (int j = 0; j < KNB; ++j) {
      int nbj = nbl[r * KNB + j];
      const float* kg = ws.kw + (size_t)nbj * 256 + h * KD;
      float dt = 0.f;
      for (int d = 0; d < KD; ++d) dt += q[d] * kg[d];
      const __bf16* pr = pairb + (r * KNB + j) * P;
      float bs = 0.f;
      for (int c = 0; c < P; ++c) bs += bf2f(pr[c]) * bw.w_bias[c * H + h];
      const float* kp = ws.kpgw + (size_t)nbj * 96 + h * 12;
      float d2 = 0.f;
      for (int i = 0; i < 12; ++i) { float df = qp[i] - kp[i]; d2 += df * df; }
      float lg = dt * 0.1767766953f + bs - sp * d2 * 0.125f + mb[r * KNB + j];
      logit[j] = lg;
      mx = fmaxf(mx, lg);
    }
    float se = 0.f;
    for (int j = 0; j < KNB; ++j) { logit[j] = __expf(logit[j] - mx); se += logit[j]; }
    float inv = 1.f / se;
    float o[KD], opair[P], og[12];
    for (int d = 0; d < KD; ++d) o[d] = 0.f;
    for (int c = 0; c < P; ++c) opair[c] = 0.f;
    for (int i = 0; i < 12; ++i) og[i] = 0.f;
    for (int j = 0; j < KNB; ++j) {
      float a = logit[j] * inv;
      int nbj = nbl[r * KNB + j];
      const float* vg = ws.vw + (size_t)nbj * 256 + h * KD;
      for (int d = 0; d < KD; ++d) o[d] += a * vg[d];
      const __bf16* pr = pairb + (r * KNB + j) * P;
      for (int c = 0; c < P; ++c) opair[c] += a * bf2f(pr[c]);
      const float* vp = ws.vpgw + (size_t)nbj * 96 + h * 12;
      for (int i = 0; i < 12; ++i) og[i] += a * vp[i];
    }
    const float* Rc = Rf + r * 12;
    __bf16* ob = outb + r * OUTF;
    for (int d = 0; d < KD; ++d) ob[h * KD + d] = f2bf(o[d]);
    for (int c = 0; c < P; ++c) ob[256 + h * P + c] = f2bf(opair[c]);
    for (int p = 0; p < NP; ++p) {
      float a0 = og[p * 3 + 0] - Rc[9], a1 = og[p * 3 + 1] - Rc[10], a2 = og[p * 3 + 2] - Rc[11];
      float l0 = Rc[0] * a0 + Rc[1] * a1 + Rc[2] * a2;
      float l1 = Rc[3] * a0 + Rc[4] * a1 + Rc[5] * a2;
      float l2 = Rc[6] * a0 + Rc[7] * a1 + Rc[8] * a2;
      ob[768 + h * 12 + p * 3 + 0] = f2bf(l0);
      ob[768 + h * 12 + p * 3 + 1] = f2bf(l1);
      ob[768 + h * 12 + p * 3 + 2] = f2bf(l2);
      ob[864 + h * 4 + p] = f2bf(sqrtf(l0 * l0 + l1 * l1 + l2 * l2 + 1e-12f));
    }
  }
  __syncthreads();

  // ---- phase 3: local += out @ w_out (k=896 -> 28 ktiles, A from LDS) -
  for (int nt = wv; nt < 16; nt += 4) {
    f32x8 c = fragCload(locf + nt * 16, D, lane);
    c = gemm_acc(c, outb, OUTF, bw.p_out, 28, nt, lane);
    fragCstore(locf + nt * 16, D, lane, c);
  }
  __syncthreads();

  // ---- phase 4: gated FFN -------------------------------------------
  __bf16* xb  = (__bf16*)scr;            // [16][256]
  __bf16* hb2 = (__bf16*)(scr + 8192);   // [16][512]
  ln_rows(locf, xb, bw.ln_m_g, bw.ln_m_b, red, stat, tid);
  {
    bf16x16 af[8];
#pragma unroll
    for (int kt = 0; kt < 8; ++kt) af[kt] = fragA(xb, D, lane, kt * 32);
    for (int nt = wv; nt < 32; nt += 4) {
      f32x8 cg = {}, cv = {};
      cg = gemm_reg<8>(af, bw.p_g, nt, lane, cg);
      cv = gemm_reg<8>(af, bw.p_v2, nt, lane, cv);
      int n = lane & 15, mo = (lane >> 4) * 8;
#pragma unroll
      for (int r = 0; r < 8; ++r)
        hb2[(mo + r) * FF + nt * 16 + n] = f2bf(geluf(cg[r]) * cv[r]);
    }
  }
  __syncthreads();
  for (int nt = wv; nt < 16; nt += 4) {
    f32x8 c = fragCload(locf + nt * 16, D, lane);
    c = gemm_acc(c, hb2, FF, bw.p_o, 16, nt, lane);
    fragCstore(locf + nt * 16, D, lane, c);
  }
  __syncthreads();

  // ---- phase 5: position update --------------------------------------
  ln_rows(locf, xb, bw.ln_u_g, bw.ln_u_b, red, stat, tid);
  if (wv == 0) {
    f32x8 c = {};
    c = gemm_acc(c, xb, D, bw.p_pos, 8, 0, lane);
    fragCstore(updl, 16, lane, c);
  }
  for (int i = tid; i < TR * D; i += TPB) ws.loc[(size_t)rt * D + i] = locf[i];
  __syncthreads();
  if (tid < TR * NA) {
    int r = tid / NA, a = tid % NA;
    const float* Rc = Rf + r * 12;
    size_t base = (size_t)(rt + r) * 15 + a * 3;
    float p0 = ws.posIn[base], p1 = ws.posIn[base + 1], p2v = ws.posIn[base + 2];
    float d0 = p0 - Rc[9], d1 = p1 - Rc[10], d2 = p2v - Rc[11];
    float l0 = Rc[0] * d0 + Rc[1] * d1 + Rc[2] * d2 + updl[r * 16 + a * 3 + 0];
    float l1 = Rc[3] * d0 + Rc[4] * d1 + Rc[5] * d2 + updl[r * 16 + a * 3 + 1];
    float l2 = Rc[6] * d0 + Rc[7] * d1 + Rc[8] * d2 + updl[r * 16 + a * 3 + 2];
    float n0 = Rc[0] * l0 + Rc[3] * l1 + Rc[6] * l2 + Rc[9];
    float n1 = Rc[1] * l0 + Rc[4] * l1 + Rc[7] * l2 + Rc[10];
    float n2 = Rc[2] * l0 + Rc[5] * l1 + Rc[8] * l2 + Rc[11];
    bool um = ws.umask[(size_t)(rt + r) * NA + a] != 0;
    ws.posOut[base]     = um ? n0 : p0;
    ws.posOut[base + 1] = um ? n1 : p1;
    ws.posOut[base + 2] = um ? n2 : p2v;
  }
}

// ---------------- Final kernel ---------------------------------------------
constexpr size_t SF_POSL = 0;                     // float[16][16]
constexpr size_t SF_RF   = SF_POSL + 1024;        // float[16][12]
constexpr size_t SF_LOCF = SF_RF + 768;           // float[16][256]
constexpr size_t SF_XB   = SF_LOCF + 16384;       // bf16 [16][256]
constexpr size_t SF_UPD  = SF_XB + 8192;          // float[16][16]
constexpr size_t SF_RED  = SF_UPD + 1024;
constexpr size_t SF_STAT = SF_RED + 1024;
constexpr size_t SF_SZ   = SF_STAT + 128;

__global__ __launch_bounds__(TPB) void finalK(const float* loc, const float* posIn,
                                              const float* lng, const float* lnb,
                                              const __bf16* p_posf,
                                              const unsigned char* umask,
                                              float* out) {
  const int tid = threadIdx.x, lane = tid & 31, wv = tid >> 5;
  const int rt = blockIdx.x * TR;
  extern __shared__ char smem[];
  float*  posl = (float*)(smem + SF_POSL);
  float*  Rf   = (float*)(smem + SF_RF);
  float*  locf = (float*)(smem + SF_LOCF);
  __bf16* xb   = (__bf16*)(smem + SF_XB);
  float*  updl = (float*)(smem + SF_UPD);
  float*  red  = (float*)(smem + SF_RED);
  float*  stat = (float*)(smem + SF_STAT);

  async_load_f32(locf, loc + (size_t)rt * D, TR * D, tid);
  for (int i = tid; i < TR * 15; i += TPB) {
    int r = i / 15, c = i % 15;
    posl[r * 16 + c] = posIn[(size_t)rt * 15 + i];
  }
  async_wait();
  __syncthreads();
  if (tid < TR) frames_calc(posl + tid * 16, Rf + tid * 12);
  ln_rows(locf, xb, lng, lnb, red, stat, tid);
  if (wv == 0) {
    f32x8 c = {};
    c = gemm_acc(c, xb, D, p_posf, 8, 0, lane);
    fragCstore(updl, 16, lane, c);
  }
  __syncthreads();
  if (tid < TR * NA) {
    int r = tid / NA, a = tid % NA;
    const float* Rc = Rf + r * 12;
    size_t base = (size_t)(rt + r) * 15 + a * 3;
    float p0 = posl[r * 16 + a * 3], p1 = posl[r * 16 + a * 3 + 1], p2v = posl[r * 16 + a * 3 + 2];
    float d0 = p0 - Rc[9], d1 = p1 - Rc[10], d2 = p2v - Rc[11];
    float l0 = Rc[0] * d0 + Rc[1] * d1 + Rc[2] * d2 + 10.f * updl[r * 16 + a * 3 + 0];
    float l1 = Rc[3] * d0 + Rc[4] * d1 + Rc[5] * d2 + 10.f * updl[r * 16 + a * 3 + 1];
    float l2 = Rc[6] * d0 + Rc[7] * d1 + Rc[8] * d2 + 10.f * updl[r * 16 + a * 3 + 2];
    float n0 = Rc[0] * l0 + Rc[3] * l1 + Rc[6] * l2 + Rc[9];
    float n1 = Rc[1] * l0 + Rc[4] * l1 + Rc[7] * l2 + Rc[10];
    float n2 = Rc[2] * l0 + Rc[5] * l1 + Rc[8] * l2 + Rc[11];
    bool um = umask[(size_t)(rt + r) * NA + a] != 0;
    out[base]     = um ? n0 : p0;
    out[base + 1] = um ? n1 : p1;
    out[base + 2] = um ? n2 : p2v;
  }
}

// =====================================================================
// Host launcher
// Param flattening assumption: inputs 0..7 = local,pos,neighbours,resi,
// chain,batch,update_mask,mask; then params in pytree order (dict keys
// alphabetical): blocks[0] leaves (31), blocks[1] leaves (31), ln_f_b,
// ln_f_g, w_pos_f.
// =====================================================================
extern "C" void kernel_launch(void* const* d_in, const int* in_sizes, int n_in,
                              void* d_out, int out_size, void* d_ws,
                              size_t ws_size, hipStream_t stream) {
  (void)in_sizes; (void)n_in; (void)out_size; (void)ws_size;
  const float* in_local = (const float*)d_in[0];
  const float* in_pos   = (const float*)d_in[1];
  const int* in_nbr     = (const int*)d_in[2];
  const int* in_resi    = (const int*)d_in[3];
  const int* in_chain   = (const int*)d_in[4];
  const unsigned char* in_umask = (const unsigned char*)d_in[6];
  const unsigned char* in_mask  = (const unsigned char*)d_in[7];

  char* wsb = (char*)d_ws;
  size_t off = 0;
  auto carve = [&](size_t bytes) -> char* {
    off = (off + 255) & ~(size_t)255;
    char* p = wsb + off;
    off += bytes;
    return p;
  };
  float* loc   = (float*)carve((size_t)NRES * D * 4);
  float* posA  = (float*)carve((size_t)NRES * 15 * 4);
  float* posB  = (float*)carve((size_t)NRES * 15 * 4);
  float* Rw    = (float*)carve((size_t)NRES * 9 * 4);
  float* tw    = (float*)carve((size_t)NRES * 3 * 4);
  float* qw    = (float*)carve((size_t)NRES * 256 * 4);
  float* kw    = (float*)carve((size_t)NRES * 256 * 4);
  float* vw    = (float*)carve((size_t)NRES * 256 * 4);
  float* qpg   = (float*)carve((size_t)NRES * 96 * 4);
  float* kpg   = (float*)carve((size_t)NRES * 96 * 4);
  float* vpg   = (float*)carve((size_t)NRES * 96 * 4);
  float* centw = (float*)carve((size_t)NRES * 64 * 4);
  float* neiw  = (float*)carve((size_t)NRES * 64 * 4);
  __bf16* packbase = (__bf16*)carve((size_t)3800 * 512 * 2);

  __bf16* pp = packbase;
  auto packto = [&](const float* W, int Ks, int Ns, int Kp, int Np) -> const __bf16* {
    int tiles = (Kp / 32) * (Np / 16);
    __bf16* dst = pp;
    pp += (size_t)tiles * 512;
    int total = tiles * 512;
    pack_weight<<<(total + 255) / 256, 256, 0, stream>>>(W, dst, Ks, Ns, Kp, Np);
    return dst;
  };

  BWT bwv[2];
  for (int b = 0; b < 2; ++b) {
    int base = 8 + b * 31;
    auto Fp = [&](int o) { return (const float*)d_in[base + o]; };
    BWT& bw = bwv[b];
    bw.b_p1 = Fp(0);   bw.gamma = Fp(1);
    bw.ln_a_b = Fp(2); bw.ln_a_g = Fp(3);
    bw.ln_m_b = Fp(4); bw.ln_m_g = Fp(5);
    bw.ln_p1_b = Fp(6); bw.ln_p1_g = Fp(7);
    bw.ln_p2_b = Fp(8); bw.ln_p2_g = Fp(9);
    bw.ln_u_b = Fp(10); bw.ln_u_g = Fp(11);
    bw.w_bias = Fp(12); bw.w_geom = Fp(15); bw.w_resi = Fp(21);
    bw.p_feat   = packto(Fp(14), 95, 256, 96, 256);
    bw.p_center = packto(Fp(13), 256, 64, 256, 64);
    bw.p_nei    = packto(Fp(16), 256, 64, 256, 64);
    bw.p_p1     = packto(Fp(18), 64, 128, 64, 128);
    bw.p_p2     = packto(Fp(19), 128, 64, 128, 64);
    bw.p_q      = packto(Fp(26), 256, 256, 256, 256);
    bw.p_k      = packto(Fp(23), 256, 256, 256, 256);
    bw.p_v      = packto(Fp(28), 256, 256, 256, 256);
    bw.p_qp     = packto(Fp(27), 256, 96, 256, 96);
    bw.p_kp     = packto(Fp(24), 256, 96, 256, 96);
    bw.p_vp     = packto(Fp(30), 256, 96, 256, 96);
    bw.p_out    = packto(Fp(17), 896, 256, 896, 256);
    bw.p_g      = packto(Fp(22), 256, 512, 256, 512);
    bw.p_v2     = packto(Fp(29), 256, 512, 256, 512);
    bw.p_o      = packto(Fp(25), 512, 256, 512, 256);
    bw.p_pos    = packto(Fp(20), 256, 15, 256, 16);
  }
  const __bf16* p_posf = packto((const float*)d_in[72], 256, 15, 256, 16);

  hipMemcpyAsync(loc, in_local, (size_t)NRES * D * 4, hipMemcpyDeviceToDevice, stream);
  hipMemcpyAsync(posA, in_pos, (size_t)NRES * 15 * 4, hipMemcpyDeviceToDevice, stream);

  WSB w{};
  w.loc = loc; w.Rw = Rw; w.tw = tw;
  w.qw = qw; w.kw = kw; w.vw = vw;
  w.qpgw = qpg; w.kpgw = kpg; w.vpgw = vpg;
  w.centw = centw; w.neiw = neiw;
  w.nbr = in_nbr; w.resi = in_resi; w.chain = in_chain;
  w.umask = in_umask; w.mask = in_mask;

  dim3 grid(NRES / TR), blk(TPB);
  // block 0: pos ping A -> B
  w.posIn = posA; w.posOut = posB;
  stageA<<<grid, blk, SA_SZ, stream>>>(w, bwv[0]);
  stageB<<<grid, blk, SB_SZ, stream>>>(w, bwv[0]);
  // block 1: pos ping B -> A
  w.posIn = posB; w.posOut = posA;
  stageA<<<grid, blk, SA_SZ, stream>>>(w, bwv[1]);
  stageB<<<grid, blk, SB_SZ, stream>>>(w, bwv[1]);
  // final projection
  finalK<<<grid, blk, SF_SZ, stream>>>(loc, posA, (const float*)d_in[71],
                                       (const float*)d_in[70], p_posf, in_umask,
                                       (float*)d_out);
}
